// SpatialPatchMotionAttention_86234353369770
// MI455X (gfx1250) — matrix operand order repaired and checked by reading, not compiled, not run on hardware
//
#include <hip/hip_runtime.h>
#include <hip/hip_bf16.h>
#include <math.h>

// ---- CDNA5 WMMA types ----
typedef __attribute__((ext_vector_type(16))) __bf16 v16bf;
typedef __attribute__((ext_vector_type(8)))  __bf16 v8bf;
typedef __attribute__((ext_vector_type(4)))  __bf16 v4bf;
typedef __attribute__((ext_vector_type(8)))  float  v8f;

// ---- problem constants ----
constexpr int kT    = 8;      // n_segment
constexpr int kC    = 256;
constexpr int kH    = 28;
constexpr int kW    = 28;
constexpr int kPW   = 7;      // patches per side
constexpr int kE    = 4096;   // patch embedding dim
constexpr int kNPOS = 49;     // ph*pw
constexpr int kB    = 16;
constexpr int kBT   = 128;

// gather one element of y[(b,ph,pw)][t][e] from the raw (bt,c,h,w) tensor
__device__ __forceinline__ float load_x(const float* __restrict__ x,
                                        int b, int t, int ph, int pw, int e) {
  int c  = e & 255;
  int p2 = (e >> 8) & 3;
  int p1 = e >> 10;
  return x[(((size_t)(b * kT + t) * kC + c) * kH + (ph * 4 + p1)) * kW + (pw * 4 + p2)];
}

__device__ __forceinline__ float wave_sum(float v) {
#pragma unroll
  for (int m = 16; m > 0; m >>= 1) v += __shfl_xor(v, m, 32);
  return v;
}

__device__ __forceinline__ v8f wmma_bf16(v16bf a, v16bf b, v8f c) {
  return __builtin_amdgcn_wmma_f32_16x16x32_bf16(false, a, false, b, (short)0, c, false, false);
}

// load a 16-element bf16 fragment as two contiguous 16B chunks (-> ds_load_b128)
__device__ __forceinline__ v16bf ld_frag2(const __bf16* p0, const __bf16* p1) {
  v8bf lo = *(const v8bf*)p0;
  v8bf hi = *(const v8bf*)p1;
  return __builtin_shufflevector(lo, hi, 0, 1, 2, 3, 4, 5, 6, 7,
                                         8, 9, 10, 11, 12, 13, 14, 15);
}
// A-fragment (16x32 bf16): chunks at  +g*16 and  +g*16+8   (K-contiguous)
// B-fragment (32x16 bf16): chunks at  +g*16 and  +g*16+8   (K-contiguous)

// ======================= Kernel 1: temporal attention =======================
// one block per (b, ph, pw); y tile is (8, 4096); 4 waves
constexpr int SROW = 4104;  // s_sh row stride: 8208B -> 16B aligned, bank shift 4
constexpr int TP_E = 512;   // e-panel width for the S@V stage
constexpr int TP_S = 24;    // sT row stride: 48B -> 16B aligned, bank shift 12

__global__ void __launch_bounds__(128)
temporal_attn_kernel(const float* __restrict__ x, float* __restrict__ z)
{
  __shared__ __attribute__((aligned(32))) __bf16 s_sh[9 * SROW];   // rows 0..7 data, row 8 = zeros
  __shared__ __attribute__((aligned(32))) __bf16 sT[TP_E * TP_S];  // transposed e-panel
  __shared__ __attribute__((aligned(32))) __bf16 Sbf[16 * 32];     // softmax(S), K-padded w/ zeros
  __shared__ __attribute__((aligned(32))) __bf16 zbuf[16];         // 32B of zeros
  __shared__ float mean_sh[kT], rstd_sh[kT];
  __shared__ float Spart[4][16][16];

  const int blk = blockIdx.x;
  const int b   = blk / kNPOS;
  const int pos = blk % kNPOS;
  const int ph  = pos / kPW;
  const int pw  = pos % kPW;

  const int tid  = threadIdx.x;
  const int wv   = tid >> 5;       // 0..3
  const int lane = tid & 31;
  const int m    = lane & 15;      // M row (A) / N col (B, C)
  const int g    = lane >> 4;      // lane-half
  const int rb   = g * 8;          // C-fragment row base

  // zero-fill the pad structures (row 8 of s_sh, Sbf, zbuf)
  for (int f = tid; f < SROW; f += 128) s_sh[8 * SROW + f] = (__bf16)0.f;
  for (int f = tid; f < 16 * 32; f += 128) Sbf[f] = (__bf16)0.f;
  if (tid < 16) zbuf[tid] = (__bf16)0.f;

  // Phase 1: per-row mean / rstd (rows = t); 2 rows per wave
  for (int j = 0; j < 2; ++j) {
    int t = wv * 2 + j;
    float s = 0.f, s2 = 0.f;
    for (int e = lane; e < kE; e += 32) {
      float v = load_x(x, b, t, ph, pw, e);
      s += v; s2 += v * v;
    }
    s = wave_sum(s); s2 = wave_sum(s2);
    if (lane == 0) {
      float mn  = s * (1.f / kE);
      float var = s2 * (1.f / kE) - mn * mn;
      mean_sh[t] = mn;
      rstd_sh[t] = rsqrtf(var + 1e-5f);
    }
  }
  __syncthreads();

  // Phase 2: s[t] = xn[t+1] - xn[t]  (xn[8] := 0), bf16 into LDS rows
  for (int f = tid; f < kT * kE; f += 128) {
    int t = f >> 12;
    int e = f & (kE - 1);
    float xn  = (load_x(x, b, t, ph, pw, e) - mean_sh[t]) * rstd_sh[t];
    float xn1 = (t < kT - 1)
                ? (load_x(x, b, t + 1, ph, pw, e) - mean_sh[t + 1]) * rstd_sh[t + 1]
                : 0.f;
    s_sh[t * SROW + e] = (__bf16)(xn1 - xn);
  }
  __syncthreads();

  // Phase 3: S_raw = s @ s^T ; each wave owns a strided slice of K.
  // Counted loop (uniform trip count) -> scalar loop control + pipelining.
  {
    const __bf16* row = s_sh + ((m < kT) ? m : 8) * SROW;  // A-row == B-row (q=k=v)
    v8f acc = {};
#pragma unroll 2
    for (int it = 0; it < kE / 128; ++it) {
      int k0 = it * 128 + wv * 32;
      v16bf a  = ld_frag2(row + k0 + g * 16, row + k0 + g * 16 + 8);
      v16bf bf = ld_frag2(row + k0 + g * 16, row + k0 + g * 16 + 8);
      acc = wmma_bf16(a, bf, acc);
    }
#pragma unroll
    for (int v = 0; v < 8; ++v) Spart[wv][rb + v][m] = acc[v];
  }
  __syncthreads();

  // Phase 4: row softmax over the 8 valid cols, scale 1/sqrt(4096) = 1/64
  if (tid < kT) {
    int r = tid;
    float raw[kT];
    float mx = -3.4e38f;
#pragma unroll
    for (int j = 0; j < kT; ++j) {
      float v = (Spart[0][r][j] + Spart[1][r][j] + Spart[2][r][j] + Spart[3][r][j]) * 0.015625f;
      raw[j] = v;
      mx = fmaxf(mx, v);
    }
    float sum = 0.f;
#pragma unroll
    for (int j = 0; j < kT; ++j) { raw[j] = __expf(raw[j] - mx); sum += raw[j]; }
    float inv = 1.f / sum;
#pragma unroll
    for (int j = 0; j < kT; ++j) Sbf[r * 32 + j] = (__bf16)(raw[j] * inv);
  }
  __syncthreads();

  // Phase 5: out = S @ s + y ; e-panels of 512, staged transposed in LDS.
  {
    // A-fragment is constant: Sbf rows, K 0..15 data / 16..31 zero-pad
    v16bf aS = ld_frag2(&Sbf[m * 32 + g * 16], &Sbf[m * 32 + g * 16 + 8]);

    for (int ep = 0; ep < kE; ep += TP_E) {
      // stage sT[e_local][t] (t >= 8 zeroed)
      for (int f = tid; f < TP_E * 16; f += 128) {
        int el = f >> 4, t = f & 15;
        sT[el * TP_S + t] = (t < kT) ? s_sh[t * SROW + ep + el] : (__bf16)0.f;
      }
      __syncthreads();

      // 32 tiles per panel, 8 per wave (counted loop)
      for (int it = 0; it < TP_E / 16 / 4; ++it) {
        int ti = it * 4 + wv;
        // B-fragment: K 0..15 from sT row, K 16..31 from the zero buffer
        const __bf16* p = (g == 0) ? &sT[(ti * 16 + m) * TP_S] : zbuf;
        v16bf bf = ld_frag2(p, p + 8);
        v8f acc = {};
        acc = wmma_bf16(aS, bf, acc);
#pragma unroll
        for (int v = 0; v < 8; ++v) {
          int t = rb + v;
          if (t < kT) {
            int e = ep + ti * 16 + m;
            float val = acc[v] + load_x(x, b, t, ph, pw, e);
            z[((size_t)(b * kT + t) * kNPOS + pos) * kE + e] = val;
          }
        }
      }
      __syncthreads();
    }
  }
}

// ======================= Kernel 2: spatial attention =======================
// one block per bt; y tile is (49, 4096) padded to 64 rows; 8 waves
constexpr int XNP_S = 136;  // xn panel row stride: 272B -> 16B aligned, bank shift 4
constexpr int XT_S  = 72;   // xn^T / Sb row stride: 144B -> 16B aligned, bank shift 36
constexpr int SS_S  = 68;   // f32 score row stride (scalar access only)

__global__ void __launch_bounds__(256)
spatial_attn_kernel(const float* __restrict__ z, float* __restrict__ out)
{
  __shared__ float mean_sh[64], rstd_sh[64];
  __shared__ __attribute__((aligned(32))) __bf16 xnp[64 * XNP_S];   // Xn K-panel (row-major)
  __shared__ __attribute__((aligned(32))) __bf16 xnT[128 * XT_S];   // Xn^T N-panel
  __shared__ __attribute__((aligned(32))) __bf16 Sb [64 * XT_S];    // softmax(S), 64 cols + pad
  __shared__ float Ssc[64 * SS_S];                                  // raw / exp scores

  const int bt = blockIdx.x;
  const size_t base = (size_t)bt * kNPOS * kE;
  const int tid  = threadIdx.x;
  const int wv   = tid >> 5;   // 0..7
  const int lane = tid & 31;
  const int m    = lane & 15;
  const int g    = lane >> 4;
  const int rb   = g * 8;

  // Phase A: per-row layernorm stats (rows >= 49 stage as zeros later)
  for (int j = 0; j < 8; ++j) {
    int r = wv * 8 + j;
    float s = 0.f, s2 = 0.f;
    if (r < kNPOS) {
      for (int e = lane; e < kE; e += 32) {
        float v = z[base + (size_t)r * kE + e];
        s += v; s2 += v * v;
      }
    }
    s = wave_sum(s); s2 = wave_sum(s2);
    if (lane == 0) {
      float mn  = s * (1.f / kE);
      float var = s2 * (1.f / kE) - mn * mn;
      mean_sh[r] = mn;
      rstd_sh[r] = (r < kNPOS) ? rsqrtf(var + 1e-5f) : 0.f;
    }
  }
  __syncthreads();

  // Phase B: S = Xn @ Xn^T (64x64), K = 4096 in 128-wide LDS panels.
  // 16 score tiles (4x4); each wave owns 2 tiles sharing one A-fragment.
  const int t0  = wv * 2;
  const int mi  = t0 >> 2;
  const int ni0 = t0 & 3;
  const __bf16* arow  = xnp + (mi * 16 + m) * XNP_S;
  const __bf16* brow0 = xnp + (ni0 * 16 + m) * XNP_S;
  const __bf16* brow1 = xnp + ((ni0 + 1) * 16 + m) * XNP_S;
  v8f acc0 = {}, acc1 = {};
  for (int kp = 0; kp < kE; kp += 128) {
    // stage: float4 global loads, packed v4bf LDS stores
    for (int f = tid; f < 64 * 32; f += 256) {
      int r = f >> 5, k4 = (f & 31) * 4;
      float4 v4 = make_float4(0.f, 0.f, 0.f, 0.f);
      if (r < kNPOS) v4 = *(const float4*)&z[base + (size_t)r * kE + kp + k4];
      float mn = mean_sh[r], rs = rstd_sh[r];
      v4bf p;
      p[0] = (__bf16)((v4.x - mn) * rs);
      p[1] = (__bf16)((v4.y - mn) * rs);
      p[2] = (__bf16)((v4.z - mn) * rs);
      p[3] = (__bf16)((v4.w - mn) * rs);
      *(v4bf*)&xnp[r * XNP_S + k4] = p;
    }
    // prefetch next K-panel (global_prefetch_b8): 64 rows x 4 cachelines
    if (kp + 128 < kE)
      __builtin_prefetch(&z[base + (size_t)(tid >> 2) * kE + kp + 128 + (tid & 3) * 32], 0, 1);
    __syncthreads();
#pragma unroll
    for (int ks = 0; ks < 128; ks += 32) {
      v16bf a  = ld_frag2(arow  + ks + g * 16, arow  + ks + g * 16 + 8);
      v16bf b0 = ld_frag2(brow0 + ks + g * 16, brow0 + ks + g * 16 + 8);
      v16bf b1 = ld_frag2(brow1 + ks + g * 16, brow1 + ks + g * 16 + 8);
      acc0 = wmma_bf16(a, b0, acc0);
      acc1 = wmma_bf16(a, b1, acc1);
    }
    __syncthreads();
  }
#pragma unroll
  for (int v = 0; v < 8; ++v) {
    Ssc[(mi * 16 + rb + v) * SS_S + ni0 * 16 + m]       = acc0[v] * 0.015625f;
    Ssc[(mi * 16 + rb + v) * SS_S + (ni0 + 1) * 16 + m] = acc1[v] * 0.015625f;
  }
  __syncthreads();

  // Phase C: row softmax over the 49 valid columns; cols 49..63 -> 0
  if (tid < 64) {
    int r = tid;
    float mx = -3.4e38f;
    for (int jc = 0; jc < kNPOS; ++jc) mx = fmaxf(mx, Ssc[r * SS_S + jc]);
    float sum = 0.f;
    for (int jc = 0; jc < kNPOS; ++jc) {
      float ev = __expf(Ssc[r * SS_S + jc] - mx);
      Ssc[r * SS_S + jc] = ev;
      sum += ev;
    }
    float inv = 1.f / sum;
    for (int jc = 0; jc < 64; ++jc)
      Sb[r * XT_S + jc] = (__bf16)((jc < kNPOS) ? Ssc[r * SS_S + jc] * inv : 0.f);
  }
  __syncthreads();

  // Phase D: out = S @ Xn + Y, scatter back to (bt, c, h, w).
  // N-panels of 128 staged transposed; 32 tiles/panel, 4 per wave, K = 64.
  const int mi2 = wv >> 1;
  const int nib = (wv & 1) * 4;
  const __bf16* sbrow = Sb + (mi2 * 16 + m) * XT_S;
  v16bf a0 = ld_frag2(sbrow + g * 16,      sbrow + g * 16 + 8);       // K 0..31
  v16bf a1 = ld_frag2(sbrow + 32 + g * 16, sbrow + 32 + g * 16 + 8);  // K 32..63
  for (int ep = 0; ep < kE; ep += 128) {
    // stage xnT[e_local][k-row] (transposed; rows >= 49 are zeros)
    for (int f = tid; f < 64 * 32; f += 256) {
      int r = f >> 5, k4 = (f & 31) * 4;
      float4 v4 = make_float4(0.f, 0.f, 0.f, 0.f);
      if (r < kNPOS) v4 = *(const float4*)&z[base + (size_t)r * kE + ep + k4];
      float mn = mean_sh[r], rs = rstd_sh[r];
      xnT[(k4 + 0) * XT_S + r] = (__bf16)((v4.x - mn) * rs);
      xnT[(k4 + 1) * XT_S + r] = (__bf16)((v4.y - mn) * rs);
      xnT[(k4 + 2) * XT_S + r] = (__bf16)((v4.z - mn) * rs);
      xnT[(k4 + 3) * XT_S + r] = (__bf16)((v4.w - mn) * rs);
    }
    if (ep + 128 < kE)
      __builtin_prefetch(&z[base + (size_t)(tid >> 2) * kE + ep + 128 + (tid & 3) * 32], 0, 1);
    __syncthreads();
#pragma unroll
    for (int tj = 0; tj < 4; ++tj) {
      int ni = nib + tj;
      const __bf16* brow = xnT + (ni * 16 + m) * XT_S;
      v16bf b0 = ld_frag2(brow + g * 16,      brow + g * 16 + 8);
      v16bf b1 = ld_frag2(brow + 32 + g * 16, brow + 32 + g * 16 + 8);
      v8f acc = {};
      acc = wmma_bf16(a0, b0, acc);
      acc = wmma_bf16(a1, b1, acc);
#pragma unroll
      for (int v = 0; v < 8; ++v) {
        int row = mi2 * 16 + rb + v;
        if (row < kNPOS) {
          int e = ep + ni * 16 + m;
          float val = acc[v] + z[base + (size_t)row * kE + e];
          int c  = e & 255, p2 = (e >> 8) & 3, p1 = e >> 10;
          int ph = row / kPW, pw = row % kPW;
          out[(((size_t)bt * kC + c) * kH + (ph * 4 + p1)) * kW + (pw * 4 + p2)] = val;
        }
      }
    }
    __syncthreads();
  }
}

// ======================= launch =======================
extern "C" void kernel_launch(void* const* d_in, const int* in_sizes, int n_in,
                              void* d_out, int out_size, void* d_ws, size_t ws_size,
                              hipStream_t stream) {
  (void)in_sizes; (void)n_in; (void)out_size; (void)ws_size;
  const float* x = (const float*)d_in[0];
  float* out = (float*)d_out;
  // intermediate z: (bt, 49, 4096) f32 = 128*49*4096*4 B ~= 98 MB (L2-resident)
  float* z = (float*)d_ws;

  temporal_attn_kernel<<<dim3(kB * kNPOS), dim3(128), 0, stream>>>(x, z);
  spatial_attn_kernel <<<dim3(kBT),        dim3(256), 0, stream>>>(z, out);
}
